// GlobalMPNNLayer_14620068675877
// MI455X (gfx1250) — matrix operand order repaired and checked by reading, not compiled, not run on hardware
//
#include <hip/hip_runtime.h>
#include <cstddef>

// Problem constants (match reference)
#define N_   20000
#define E_   640000
#define G_   256
#define HD   128
#define ED   64
#define GD   64
#define HID  128

typedef __attribute__((ext_vector_type(16))) __bf16 v16bf;
typedef __attribute__((ext_vector_type(8)))  float  v8f;

// ---------------------------------------------------------------------------
// WMMA helpers (CDNA5 wave32, V_WMMA_F32_16X16X32_BF16)
//
// A fragment (16x32 bf16, MxK): lane L: m = L&15, half = L>>4
//   element e<8  -> K = 8*half + e          (consecutive in memory)
//   element e>=8 -> K = 16 + 8*half + (e-8) (consecutive in memory)
// B fragment (32x16 bf16, KxN): lane L: n = L&15, K same as A fragment
//   (weights pre-swizzled so each lane's 16 elements are contiguous 32B)
// C/D (16x16 f32): vec elem j: m = j + 8*half, n = L&15
// ---------------------------------------------------------------------------

__device__ __forceinline__ v8f wmma_bf16(v16bf a, v16bf b, v8f c) {
  return __builtin_amdgcn_wmma_f32_16x16x32_bf16(
      false, a, false, b, (short)0, c, false, false);
}

// Build A fragment from f32 memory. p = rowBase + koff + 8*half
__device__ __forceinline__ v16bf a_from_f32(const float* __restrict__ p) {
  v16bf a;
#pragma unroll
  for (int i = 0; i < 8; ++i) {
    a[i]     = (__bf16)p[i];
    a[i + 8] = (__bf16)p[16 + i];
  }
  return a;
}

// Build A fragment from sum of two f32 arrays (m + g_map style)
__device__ __forceinline__ v16bf a_from_sum(const float* __restrict__ p,
                                            const float* __restrict__ q) {
  v16bf a;
#pragma unroll
  for (int i = 0; i < 8; ++i) {
    a[i]     = (__bf16)(p[i] + q[i]);
    a[i + 8] = (__bf16)(p[16 + i] + q[16 + i]);
  }
  return a;
}

// Build A fragment from a bf16 LDS tile. p = rowBase + koff + 8*half
__device__ __forceinline__ v16bf a_from_lds(const __bf16* p) {
  v16bf a;
#pragma unroll
  for (int i = 0; i < 8; ++i) {
    a[i]     = p[i];
    a[i + 8] = p[16 + i];
  }
  return a;
}

__device__ __forceinline__ v16bf b_load(const __bf16* __restrict__ w) {
  return *(const v16bf*)w;
}

// ---------------------------------------------------------------------------
// Weight swizzle: W[K,N] f32 (row-major) -> bf16 B-fragment stream
// out[(((kt*NT + nt)*32 + lane)*16) + i] = W[kt*32 + (i<8 ? 8*half+i : 16+8*half+i-8)][nt*16 + l]
// ---------------------------------------------------------------------------
__global__ void k_swz(const float* __restrict__ W, __bf16* __restrict__ out,
                      int Kdim, int Ndim) {
  int idx = blockIdx.x * blockDim.x + threadIdx.x;
  if (idx >= Kdim * Ndim) return;
  int i    = idx & 15;
  int lane = (idx >> 4) & 31;
  int t    = idx >> 9;
  int NT   = Ndim >> 4;
  int nt   = t % NT;
  int kt   = t / NT;
  int hf   = lane >> 4, l = lane & 15;
  int kk   = kt * 32 + (i < 8 ? 8 * hf + i : 16 + 8 * hf + (i - 8));
  int n    = nt * 16 + l;
  out[idx] = (__bf16)W[kk * Ndim + n];
}

// ---------------------------------------------------------------------------
// g_map = g @ Wmap + bmap   [G, HID], K=GD (tiny)
// ---------------------------------------------------------------------------
__global__ void k_gmap(const float* __restrict__ g, const float* __restrict__ W,
                       const float* __restrict__ b, float* __restrict__ out) {
  int idx = blockIdx.x * blockDim.x + threadIdx.x;
  if (idx >= G_ * HID) return;
  int gi = idx >> 7, c = idx & 127;
  float s = b[c];
  for (int k = 0; k < GD; ++k) s += g[gi * GD + k] * W[k * HID + c];
  out[idx] = s;
}

// ---------------------------------------------------------------------------
// Edge message: m = relu(relu([h[dst],h[src],e] @ Wm1 + bm1) @ Wm2 + bm2)
// Two 16-row M-tiles per wave (B fragments reused for both) to halve the
// L2 weight-stream traffic. K1=320 (10 steps), N1=128; K2=128 (4), N2=128.
// ---------------------------------------------------------------------------
__global__ __launch_bounds__(128) void k_edge_msg(
    const float* __restrict__ h, const float* __restrict__ e,
    const int* __restrict__ eidx, const __bf16* __restrict__ w1,
    const __bf16* __restrict__ w2, const float* __restrict__ bm1,
    const float* __restrict__ bm2, float* __restrict__ mbuf) {
  __shared__ __bf16 lds[4][2][16 * HID];  // 32 KB / workgroup
  const int wave = threadIdx.x >> 5, lane = threadIdx.x & 31;
  const int pair = blockIdx.x * 4 + wave;  // E_/32 pairs, exact
  const int t0 = pair * 2;
  const int l = lane & 15, hf = lane >> 4;
  const int r0 = t0 * 16 + l, r1 = r0 + 16;
  const int s0 = eidx[r0], d0 = eidx[E_ + r0];
  const int s1 = eidx[r1], d1 = eidx[E_ + r1];

  v8f acc[2][8] = {};
#pragma unroll
  for (int kt = 0; kt < 10; ++kt) {
    const float *p0, *p1;
    if (kt < 4) {
      p0 = h + (size_t)d0 * HD + kt * 32;
      p1 = h + (size_t)d1 * HD + kt * 32;
    } else if (kt < 8) {
      p0 = h + (size_t)s0 * HD + (kt - 4) * 32;
      p1 = h + (size_t)s1 * HD + (kt - 4) * 32;
    } else {
      p0 = e + (size_t)r0 * ED + (kt - 8) * 32;
      p1 = e + (size_t)r1 * ED + (kt - 8) * 32;
    }
    v16bf a0 = a_from_f32(p0 + 8 * hf);
    v16bf a1 = a_from_f32(p1 + 8 * hf);
#pragma unroll
    for (int nt = 0; nt < 8; ++nt) {
      v16bf b = b_load(w1 + ((size_t)(kt * 8 + nt) * 32 + lane) * 16);
      acc[0][nt] = wmma_bf16(a0, b, acc[0][nt]);
      acc[1][nt] = wmma_bf16(a1, b, acc[1][nt]);
    }
  }
#pragma unroll
  for (int mt = 0; mt < 2; ++mt) {
    __bf16* t = lds[wave][mt];
#pragma unroll
    for (int nt = 0; nt < 8; ++nt) {
      const int n = nt * 16 + l;
      const float bias = bm1[n];
#pragma unroll
      for (int j = 0; j < 8; ++j) {
        float v = acc[mt][nt][j] + bias;
        t[(j + 8 * hf) * HID + n] = (__bf16)(v > 0.f ? v : 0.f);
      }
    }
  }
  v8f acc2[2][8] = {};
#pragma unroll
  for (int kt = 0; kt < 4; ++kt) {
    v16bf a0 = a_from_lds(lds[wave][0] + l * HID + kt * 32 + 8 * hf);
    v16bf a1 = a_from_lds(lds[wave][1] + l * HID + kt * 32 + 8 * hf);
#pragma unroll
    for (int nt = 0; nt < 8; ++nt) {
      v16bf b = b_load(w2 + ((size_t)(kt * 8 + nt) * 32 + lane) * 16);
      acc2[0][nt] = wmma_bf16(a0, b, acc2[0][nt]);
      acc2[1][nt] = wmma_bf16(a1, b, acc2[1][nt]);
    }
  }
#pragma unroll
  for (int mt = 0; mt < 2; ++mt) {
    float* mr = mbuf + (size_t)(t0 + mt) * 16 * HID;
#pragma unroll
    for (int nt = 0; nt < 8; ++nt) {
      const int n = nt * 16 + l;
      const float bias = bm2[n];
#pragma unroll
      for (int j = 0; j < 8; ++j) {
        float v = acc2[mt][nt][j] + bias;
        mr[(j + 8 * hf) * HID + n] = v > 0.f ? v : 0.f;
      }
    }
  }
}

// ---------------------------------------------------------------------------
// Edge update: e_out = relu(e + relu([e, m+gmap[batch[dst]]]@We1+be1)@We2+be2)
// Two M-tiles per wave. K1=192 (6 steps), N1=64; K2=64 (2), N2=64.
// ---------------------------------------------------------------------------
__global__ __launch_bounds__(128) void k_edge_upd(
    const float* __restrict__ e, const int* __restrict__ eidx,
    const int* __restrict__ batch, const float* __restrict__ mbuf,
    const float* __restrict__ gmap, const __bf16* __restrict__ w1,
    const __bf16* __restrict__ w2, const float* __restrict__ be1,
    const float* __restrict__ be2, float* __restrict__ eout) {
  __shared__ __bf16 lds[4][2][16 * ED];  // 16 KB / workgroup
  const int wave = threadIdx.x >> 5, lane = threadIdx.x & 31;
  const int pair = blockIdx.x * 4 + wave;
  const int t0 = pair * 2;
  const int l = lane & 15, hf = lane >> 4;
  const int r0 = t0 * 16 + l, r1 = r0 + 16;
  const int d0 = eidx[E_ + r0], d1 = eidx[E_ + r1];
  const int bg0 = batch[d0], bg1 = batch[d1];
  const float* mp0 = mbuf + (size_t)r0 * HID;
  const float* gp0 = gmap + (size_t)bg0 * HID;
  const float* mp1 = mbuf + (size_t)r1 * HID;
  const float* gp1 = gmap + (size_t)bg1 * HID;

  v8f acc[2][4] = {};
#pragma unroll
  for (int kt = 0; kt < 6; ++kt) {
    v16bf a0, a1;
    if (kt < 2) {
      a0 = a_from_f32(e + (size_t)r0 * ED + kt * 32 + 8 * hf);
      a1 = a_from_f32(e + (size_t)r1 * ED + kt * 32 + 8 * hf);
    } else {
      int ko = (kt - 2) * 32 + 8 * hf;
      a0 = a_from_sum(mp0 + ko, gp0 + ko);
      a1 = a_from_sum(mp1 + ko, gp1 + ko);
    }
#pragma unroll
    for (int nt = 0; nt < 4; ++nt) {
      v16bf b = b_load(w1 + ((size_t)(kt * 4 + nt) * 32 + lane) * 16);
      acc[0][nt] = wmma_bf16(a0, b, acc[0][nt]);
      acc[1][nt] = wmma_bf16(a1, b, acc[1][nt]);
    }
  }
#pragma unroll
  for (int mt = 0; mt < 2; ++mt) {
    __bf16* t = lds[wave][mt];
#pragma unroll
    for (int nt = 0; nt < 4; ++nt) {
      const int n = nt * 16 + l;
      const float bias = be1[n];
#pragma unroll
      for (int j = 0; j < 8; ++j) {
        float v = acc[mt][nt][j] + bias;
        t[(j + 8 * hf) * ED + n] = (__bf16)(v > 0.f ? v : 0.f);
      }
    }
  }
  v8f acc2[2][4] = {};
#pragma unroll
  for (int kt = 0; kt < 2; ++kt) {
    v16bf a0 = a_from_lds(lds[wave][0] + l * ED + kt * 32 + 8 * hf);
    v16bf a1 = a_from_lds(lds[wave][1] + l * ED + kt * 32 + 8 * hf);
#pragma unroll
    for (int nt = 0; nt < 4; ++nt) {
      v16bf b = b_load(w2 + ((size_t)(kt * 4 + nt) * 32 + lane) * 16);
      acc2[0][nt] = wmma_bf16(a0, b, acc2[0][nt]);
      acc2[1][nt] = wmma_bf16(a1, b, acc2[1][nt]);
    }
  }
#pragma unroll
  for (int mt = 0; mt < 2; ++mt) {
#pragma unroll
    for (int nt = 0; nt < 4; ++nt) {
      const int n = nt * 16 + l;
      const float bias = be2[n];
#pragma unroll
      for (int j = 0; j < 8; ++j) {
        int r2 = (t0 + mt) * 16 + j + 8 * hf;
        size_t o = (size_t)r2 * ED + n;
        float v = acc2[mt][nt][j] + bias + e[o];
        eout[o] = v > 0.f ? v : 0.f;
      }
    }
  }
}

// ---------------------------------------------------------------------------
// Node update: h_out = relu(h + relu([h, m_aggr+gmap[batch]]@Wh1+bh1)@Wh2+bh2)
// K1=256 (8 steps), N1=128; K2=128 (4), N2=128. 1250 tiles (tiny workload).
// ---------------------------------------------------------------------------
__global__ __launch_bounds__(128) void k_node_upd(
    const float* __restrict__ h, const int* __restrict__ batch,
    const float* __restrict__ maggr, const float* __restrict__ gmap,
    const __bf16* __restrict__ w1, const __bf16* __restrict__ w2,
    const float* __restrict__ bh1, const float* __restrict__ bh2,
    float* __restrict__ hout) {
  __shared__ __bf16 lds[4][16 * HID];
  const int wave = threadIdx.x >> 5, lane = threadIdx.x & 31;
  const int tile = blockIdx.x * 4 + wave;
  if (tile >= N_ / 16) return;  // wave-uniform exit, EXEC stays all-ones
  const int l = lane & 15, hf = lane >> 4;
  const int row = tile * 16 + l;
  const int bn = batch[row];
  const float* mp = maggr + (size_t)row * HID;
  const float* gp = gmap + (size_t)bn * HID;

  v8f acc[8] = {};
#pragma unroll
  for (int kt = 0; kt < 8; ++kt) {
    v16bf a;
    if (kt < 4) {
      a = a_from_f32(h + (size_t)row * HD + kt * 32 + 8 * hf);
    } else {
      int ko = (kt - 4) * 32 + 8 * hf;
      a = a_from_sum(mp + ko, gp + ko);
    }
#pragma unroll
    for (int nt = 0; nt < 8; ++nt) {
      v16bf b = b_load(w1 + ((size_t)(kt * 8 + nt) * 32 + lane) * 16);
      acc[nt] = wmma_bf16(a, b, acc[nt]);
    }
  }
  __bf16* t = lds[wave];
#pragma unroll
  for (int nt = 0; nt < 8; ++nt) {
    const int n = nt * 16 + l;
    const float bias = bh1[n];
#pragma unroll
    for (int j = 0; j < 8; ++j) {
      float v = acc[nt][j] + bias;
      t[(j + 8 * hf) * HID + n] = (__bf16)(v > 0.f ? v : 0.f);
    }
  }
  v8f acc2[8] = {};
#pragma unroll
  for (int kt = 0; kt < 4; ++kt) {
    v16bf a = a_from_lds(t + l * HID + kt * 32 + 8 * hf);
#pragma unroll
    for (int nt = 0; nt < 8; ++nt) {
      v16bf b = b_load(w2 + ((size_t)(kt * 8 + nt) * 32 + lane) * 16);
      acc2[nt] = wmma_bf16(a, b, acc2[nt]);
    }
  }
#pragma unroll
  for (int nt = 0; nt < 8; ++nt) {
    const int n = nt * 16 + l;
    const float bias = bh2[n];
#pragma unroll
    for (int j = 0; j < 8; ++j) {
      int r2 = tile * 16 + j + 8 * hf;
      size_t o = (size_t)r2 * HD + n;
      float v = acc2[nt][j] + bias + h[o];
      hout[o] = v > 0.f ? v : 0.f;
    }
  }
}

// ---------------------------------------------------------------------------
// Segment softmax plumbing (edges unsorted by dst -> atomics)
// ---------------------------------------------------------------------------
__global__ void k_fill(float* __restrict__ p, size_t n, float v) {
  size_t i = (size_t)blockIdx.x * blockDim.x + threadIdx.x;
  if (i < n) p[i] = v;
}

__device__ __forceinline__ void atomicMaxF(float* addr, float val) {
  unsigned int* ua = (unsigned int*)addr;
  unsigned int old = *ua;
  while (true) {
    float cur = __uint_as_float(old);
    if (cur >= val) break;
    unsigned int assumed = old;
    old = atomicCAS(ua, assumed, __float_as_uint(val));
    if (old == assumed) break;
  }
}

__device__ __forceinline__ int seg_of(int r, const int* __restrict__ i1,
                                      const int* __restrict__ i2) {
  int s = i1[r];
  if (i2) s = i2[s];
  return s;
}

// att[r] = dot(X[r,:], w) + bias; atomicMax into segmax[seg]. Wave per row.
__global__ __launch_bounds__(128) void k_att(
    const float* __restrict__ X, const float* __restrict__ w,
    const float* __restrict__ bias, int rows, int K,
    const int* __restrict__ i1, const int* __restrict__ i2,
    float* __restrict__ att, float* __restrict__ segmax) {
  const int wave = threadIdx.x >> 5, lane = threadIdx.x & 31;
  const int r = blockIdx.x * 4 + wave;
  if (r >= rows) return;
  float s = 0.f;
  for (int k = lane; k < K; k += 32) s += X[(size_t)r * K + k] * w[k];
#pragma unroll
  for (int off = 16; off; off >>= 1) s += __shfl_xor(s, off, 32);
  if (lane == 0) {
    s += bias[0];
    att[r] = s;
    atomicMaxF(&segmax[seg_of(r, i1, i2)], s);
  }
}

__global__ void k_soft2(const float* __restrict__ att, int rows,
                        const int* __restrict__ i1, const int* __restrict__ i2,
                        const float* __restrict__ segmax,
                        float* __restrict__ segsum, float* __restrict__ ex) {
  int r = blockIdx.x * blockDim.x + threadIdx.x;
  if (r >= rows) return;
  int sg = seg_of(r, i1, i2);
  float v = __expf(att[r] - segmax[sg]);
  ex[r] = v;
  atomicAdd(&segsum[sg], v);
}

__global__ void k_aggr(const float* __restrict__ X, int rows, int C,
                       const int* __restrict__ i1, const int* __restrict__ i2,
                       const float* __restrict__ ex,
                       const float* __restrict__ segsum,
                       float* __restrict__ aggr) {
  int idx = blockIdx.x * blockDim.x + threadIdx.x;
  if (idx >= rows * C) return;
  int r = idx / C, c = idx % C;
  int sg = seg_of(r, i1, i2);
  float p = ex[r] / (segsum[sg] + 1e-16f);
  atomicAdd(&aggr[(size_t)sg * C + c], X[(size_t)r * C + c] * p);
}

// ---------------------------------------------------------------------------
// Graph MLP (tiny, scalar)
// ---------------------------------------------------------------------------
__global__ void k_g_t(const float* __restrict__ g, const float* __restrict__ ha,
                      const float* __restrict__ ea, const float* __restrict__ W,
                      const float* __restrict__ b, float* __restrict__ t) {
  int idx = blockIdx.x * blockDim.x + threadIdx.x;
  if (idx >= G_ * GD) return;
  int gi = idx >> 6, c = idx & 63;
  float s = b[c];
  for (int k = 0; k < GD + HD + ED; ++k) {
    float v;
    if (k < GD)           v = g[gi * GD + k];
    else if (k < GD + HD) v = ha[gi * HD + (k - GD)];
    else                  v = ea[gi * ED + (k - GD - HD)];
    s += v * W[k * GD + c];
  }
  t[idx] = s > 0.f ? s : 0.f;
}

__global__ void k_g_out(const float* __restrict__ g, const float* __restrict__ t,
                        const float* __restrict__ W, const float* __restrict__ b,
                        float* __restrict__ out) {
  int idx = blockIdx.x * blockDim.x + threadIdx.x;
  if (idx >= G_ * GD) return;
  int gi = idx >> 6, c = idx & 63;
  float s = b[c];
  for (int k = 0; k < GD; ++k) s += t[gi * GD + k] * W[k * GD + c];
  float v = g[idx] + s;
  out[idx] = v > 0.f ? v : 0.f;
}

// ---------------------------------------------------------------------------
extern "C" void kernel_launch(void* const* d_in, const int* in_sizes, int n_in,
                              void* d_out, int out_size, void* d_ws,
                              size_t ws_size, hipStream_t stream) {
  const float* h    = (const float*)d_in[0];
  const float* e    = (const float*)d_in[1];
  const float* g    = (const float*)d_in[2];
  const int*   eidx = (const int*)d_in[3];
  const int*   batch= (const int*)d_in[4];
  const float* Wmap = (const float*)d_in[5];  const float* bmap = (const float*)d_in[6];
  const float* Wm1  = (const float*)d_in[7];  const float* bm1  = (const float*)d_in[8];
  const float* Wm2  = (const float*)d_in[9];  const float* bm2  = (const float*)d_in[10];
  const float* Wma  = (const float*)d_in[11]; const float* bma  = (const float*)d_in[12];
  const float* Wh1  = (const float*)d_in[13]; const float* bh1  = (const float*)d_in[14];
  const float* Wh2  = (const float*)d_in[15]; const float* bh2  = (const float*)d_in[16];
  const float* Wha  = (const float*)d_in[17]; const float* bha  = (const float*)d_in[18];
  const float* We1  = (const float*)d_in[19]; const float* be1  = (const float*)d_in[20];
  const float* We2  = (const float*)d_in[21]; const float* be2  = (const float*)d_in[22];
  const float* Wea  = (const float*)d_in[23]; const float* bea  = (const float*)d_in[24];
  const float* Wg1  = (const float*)d_in[25]; const float* bg1  = (const float*)d_in[26];
  const float* Wg2  = (const float*)d_in[27]; const float* bg2  = (const float*)d_in[28];
  (void)in_sizes; (void)n_in; (void)out_size; (void)ws_size;

  float* hout = (float*)d_out;
  float* eout = hout + (size_t)N_ * HD;
  float* gout = eout + (size_t)E_ * ED;

  // Workspace carve-up (~350 MB total)
  char* ws = (char*)d_ws;
  size_t off = 0;
  auto carve = [&](size_t bytes) -> char* {
    char* p = ws + off;
    off = (off + bytes + 255) & ~(size_t)255;
    return p;
  };
  __bf16* wsw  = (__bf16*)carve(122880 * 2);  // all swizzled weights
  __bf16* wm1s = wsw + 0;
  __bf16* wm2s = wsw + 40960;
  __bf16* we1s = wsw + 57344;
  __bf16* we2s = wsw + 69632;
  __bf16* wh1s = wsw + 73728;
  __bf16* wh2s = wsw + 106496;
  float* gmap = (float*)carve((size_t)G_ * HID * 4);
  float* mbuf = (float*)carve((size_t)E_ * HID * 4);
  float* att  = (float*)carve((size_t)E_ * 4);
  float* exb  = (float*)carve((size_t)E_ * 4);
  float* eatt = (float*)carve((size_t)E_ * 4);
  float* eex  = (float*)carve((size_t)E_ * 4);
  float* hatt = (float*)carve((size_t)N_ * 4);
  float* hex  = (float*)carve((size_t)N_ * 4);
  const size_t zn = (size_t)N_ + (size_t)N_ * HID + G_ + (size_t)G_ * HID +
                    G_ + (size_t)G_ * ED;
  float* zblk  = (float*)carve(zn * 4);
  float* nsum  = zblk;
  float* maggr = nsum + N_;
  float* gsumh = maggr + (size_t)N_ * HID;
  float* haggr = gsumh + G_;
  float* gsume = haggr + (size_t)G_ * HID;
  float* eaggr = gsume + G_;
  const size_t nn = (size_t)N_ + 2 * G_;
  float* nblk = (float*)carve(nn * 4);
  float* nmx  = nblk;
  float* gmxh = nblk + N_;
  float* gmxe = gmxh + G_;
  float* gt   = (float*)carve((size_t)G_ * GD * 4);

  const int TB = 256;
  const float NEGINF = -__builtin_huge_valf();

  // init
  k_fill<<<(unsigned)((zn + TB - 1) / TB), TB, 0, stream>>>(zblk, zn, 0.f);
  k_fill<<<(unsigned)((nn + TB - 1) / TB), TB, 0, stream>>>(nblk, nn, NEGINF);

  // weight swizzles
  k_swz<<<(40960 + TB - 1) / TB, TB, 0, stream>>>(Wm1, wm1s, 320, 128);
  k_swz<<<(16384 + TB - 1) / TB, TB, 0, stream>>>(Wm2, wm2s, 128, 128);
  k_swz<<<(12288 + TB - 1) / TB, TB, 0, stream>>>(We1, we1s, 192, 64);
  k_swz<<<( 4096 + TB - 1) / TB, TB, 0, stream>>>(We2, we2s, 64, 64);
  k_swz<<<(32768 + TB - 1) / TB, TB, 0, stream>>>(Wh1, wh1s, 256, 128);
  k_swz<<<(16384 + TB - 1) / TB, TB, 0, stream>>>(Wh2, wh2s, 128, 128);

  k_gmap<<<(G_ * HID + TB - 1) / TB, TB, 0, stream>>>(g, Wmap, bmap, gmap);

  // edge message MLP (WMMA, 2 M-tiles per wave): E_/32 wave-units / 4 waves
  k_edge_msg<<<E_ / 32 / 4, 128, 0, stream>>>(h, e, eidx, wm1s, wm2s, bm1, bm2,
                                              mbuf);

  // m softmax over dst + aggregation
  k_att<<<(E_ + 3) / 4, 128, 0, stream>>>(mbuf, Wma, bma, E_, HID, eidx + E_,
                                          (const int*)nullptr, att, nmx);
  k_soft2<<<(E_ + TB - 1) / TB, TB, 0, stream>>>(att, E_, eidx + E_,
                                                 (const int*)nullptr, nmx, nsum,
                                                 exb);
  k_aggr<<<(E_ * HID + TB - 1) / TB, TB, 0, stream>>>(
      mbuf, E_, HID, eidx + E_, (const int*)nullptr, exb, nsum, maggr);

  // edge update MLP (WMMA, 2 M-tiles per wave) -> e_out
  k_edge_upd<<<E_ / 32 / 4, 128, 0, stream>>>(e, eidx, batch, mbuf, gmap, we1s,
                                              we2s, be1, be2, eout);

  // node update MLP (WMMA) -> h_out
  k_node_upd<<<(N_ / 16 + 3) / 4, 128, 0, stream>>>(h, batch, maggr, gmap,
                                                    wh1s, wh2s, bh1, bh2, hout);

  // h -> g aggregation
  k_att<<<(N_ + 3) / 4, 128, 0, stream>>>(h, Wha, bha, N_, HD, batch,
                                          (const int*)nullptr, hatt, gmxh);
  k_soft2<<<(N_ + TB - 1) / TB, TB, 0, stream>>>(hatt, N_, batch,
                                                 (const int*)nullptr, gmxh,
                                                 gsumh, hex);
  k_aggr<<<(N_ * HD + TB - 1) / TB, TB, 0, stream>>>(
      h, N_, HD, batch, (const int*)nullptr, hex, gsumh, haggr);

  // e -> g aggregation (seg = batch[dst])
  k_att<<<(E_ + 3) / 4, 128, 0, stream>>>(e, Wea, bea, E_, ED, eidx + E_, batch,
                                          eatt, gmxe);
  k_soft2<<<(E_ + TB - 1) / TB, TB, 0, stream>>>(eatt, E_, eidx + E_, batch,
                                                 gmxe, gsume, eex);
  k_aggr<<<(E_ * ED + TB - 1) / TB, TB, 0, stream>>>(e, E_, ED, eidx + E_,
                                                     batch, eex, gsume, eaggr);

  // graph MLP -> g_out
  k_g_t<<<(G_ * GD + TB - 1) / TB, TB, 0, stream>>>(g, haggr, eaggr, Wg1, bg1,
                                                    gt);
  k_g_out<<<(G_ * GD + TB - 1) / TB, TB, 0, stream>>>(g, gt, Wg2, bg2, gout);
}
